// TCF3DUnitaryPerformerModel_65317862638080
// MI455X (gfx1250) — matrix operand order repaired and checked by reading, not compiled, not run on hardware
//
#include <hip/hip_runtime.h>
#include <hip/hip_bf16.h>
#include <math.h>

// ---------------------------------------------------------------------------
// TCF3D Unitary Performer model for MI455X (gfx1250, wave32, WMMA).
// All dense GEMMs go through v_wmma_f32_16x16x32_bf16 (f32 accumulate).
// ---------------------------------------------------------------------------

typedef __attribute__((ext_vector_type(16))) __bf16        v16bf;
typedef __attribute__((ext_vector_type(8)))  float         v8f;
typedef __attribute__((ext_vector_type(4)))  unsigned int  uiv4;

#define NTOK   16384     // B * NX * NY
#define NPIX   4096      // NX * NY
#define DIMC   192
#define INNERC 512
#define FFC    768
#define MFEAT  266
#define MLD    272       // padded leading dim for feature buffers
#define DHC    64
#define DEPTHC 6
#define BATCH  4

#define DN_CONST    0.35355339059327f   // 64^-0.25
#define DN2_CONST   0.125f              // 64^-0.5
#define RATIO_CONST 0.06131393394849f   // 266^-0.5
#define EPS_K       1e-4f
#define EPS_SN      1e-5f
#define EPS_GN      1e-5f

static __device__ __forceinline__ float gelu_exact(float x) {
  return 0.5f * x * (1.0f + erff(x * 0.70710678118654752f));
}

// Pack two f32 into bf16x2 (round-half-up) with a single v_perm_b32:
// take high halves of (bits + 0x8000).  3 VALU ops per pair.
static __device__ __forceinline__ unsigned int pack2_bf16(float a, float b) {
  unsigned int ua = __float_as_uint(a) + 0x8000u;
  unsigned int ub = __float_as_uint(b) + 0x8000u;
  // result bytes: [0]=ua.b2 [1]=ua.b3 [2]=ub.b2 [3]=ub.b3
  return __builtin_amdgcn_perm(ub, ua, 0x07060302u);
}

static __device__ __forceinline__ int imin(int a, int b) { return a < b ? a : b; }

// ---------------------------------------------------------------------------
// Universal WMMA GEMM:  C[row,col] = sum_k A[row,k] * B[k,col]  (+ epilogue)
//   A[row,k]  = A[row*a_rs + k*a_ks]  (scaled by a_scale before bf16 convert)
//   B[k,col]  = B[k*b_ks + col*b_cs]
//   epilogue: +bias[col], /rowdiv[row], gelu, +resid[row*r_rs+col]
// Block tile 128x64, K-step 32, 256 threads (8 waves, 4 WMMA tiles each).
//
// Staging is two-phase: (1) issue ALL 12 global loads into a register array
// so the memory pipe keeps them in flight together, (2) convert + store to
// LDS.  ALIGNED instantiation additionally drops every bounds check and uses
// float2 (b64) loads with next-tile prefetch.
// ---------------------------------------------------------------------------
#define BMT 128
#define BNT 64
#define BKT 32
#define LDSP 40   // padded LDS row pitch (ushorts)

template <bool ALIGNED>
__global__ __launch_bounds__(256) void wmma_gemm_kernel(
    const float* __restrict__ A, long a_rs, long a_ks, float a_scale,
    const float* __restrict__ B, long b_ks, long b_cs,
    float* __restrict__ C, long c_rs,
    int Mr, int Nc, int K,
    const float* __restrict__ bias,
    const float* __restrict__ resid, long r_rs,
    const float* __restrict__ rowdiv,
    int act)
{
  __shared__ __align__(16) unsigned short As[BMT * LDSP];
  __shared__ __align__(16) unsigned short Bs[BNT * LDSP];   // stored transposed: Bs[col][k]

  const int tid  = threadIdx.x;
  const int wave = tid >> 5;
  const int lane = tid & 31;
  const int g    = lane >> 4;   // lane group (0/1)
  const int L    = lane & 15;

  const int m0 = blockIdx.y * BMT;
  const int n0 = blockIdx.x * BNT;

  const int wm2 = (wave & 3) << 1;   // wave's first M-tile (0,2,4,6)
  const int wn2 = (wave >> 2) << 1;  // wave's first N-tile (0,2)

  v8f acc[2][2] = {};

  // per-thread staging coordinates (fixed across K loop)
  const int ar = tid >> 4;             // 0..15 block-row within staging sweep
  const int kp = (tid & 15) << 1;      // K pair base 0,2,..,30

  for (int k0 = 0; k0 < K; k0 += BKT) {
    float2 va[8], vb[4];
    if constexpr (ALIGNED) {
      // ---- phase 1: issue all global loads (independent, stay in flight) ----
#pragma unroll
      for (int i = 0; i < 8; ++i) {
        int r = (i << 4) + ar;         // 0..127
        va[i] = *(const float2*)(A + (long)(m0 + r) * a_rs + (k0 + kp));
      }
#pragma unroll
      for (int i = 0; i < 4; ++i) {
        int cc = (i << 4) + ar;        // 0..63
        vb[i] = *(const float2*)(B + (long)(n0 + cc) * b_cs + (k0 + kp));
      }
      // prefetch next K-block (global_prefetch_b8)
      if (k0 + BKT < K) {
        __builtin_prefetch(A + (long)(m0 + (tid >> 1)) * a_rs + (k0 + BKT) + (tid & 1) * 16, 0, 1);
        __builtin_prefetch(B + (long)(n0 + (tid >> 2)) * b_cs + (k0 + BKT) + (tid & 3) * 8, 0, 1);
      }
    } else {
      // ---- generic path: clamped addresses + select (branch-free) ----
#pragma unroll
      for (int i = 0; i < 8; ++i) {
        int gr  = m0 + (i << 4) + ar;
        int gk0 = k0 + kp;
        int cr  = imin(gr,  Mr - 1);
        int ck0 = imin(gk0, K - 1);
        int ck1 = imin(gk0 + 1, K - 1);
        va[i].x = A[(long)cr * a_rs + (long)ck0 * a_ks];
        va[i].y = A[(long)cr * a_rs + (long)ck1 * a_ks];
      }
#pragma unroll
      for (int i = 0; i < 4; ++i) {
        int gc  = n0 + (i << 4) + ar;
        int gk0 = k0 + kp;
        int cc2 = imin(gc,  Nc - 1);
        int ck0 = imin(gk0, K - 1);
        int ck1 = imin(gk0 + 1, K - 1);
        vb[i].x = B[(long)ck0 * b_ks + (long)cc2 * b_cs];
        vb[i].y = B[(long)ck1 * b_ks + (long)cc2 * b_cs];
      }
      // zero out-of-range elements (cndmask, no control flow)
#pragma unroll
      for (int i = 0; i < 8; ++i) {
        int gr = m0 + (i << 4) + ar;
        va[i].x = (gr < Mr && k0 + kp     < K) ? va[i].x : 0.0f;
        va[i].y = (gr < Mr && k0 + kp + 1 < K) ? va[i].y : 0.0f;
      }
#pragma unroll
      for (int i = 0; i < 4; ++i) {
        int gc = n0 + (i << 4) + ar;
        vb[i].x = (gc < Nc && k0 + kp     < K) ? vb[i].x : 0.0f;
        vb[i].y = (gc < Nc && k0 + kp + 1 < K) ? vb[i].y : 0.0f;
      }
    }

    // ---- phase 2: convert + store to LDS ----
#pragma unroll
    for (int i = 0; i < 8; ++i) {
      int r = (i << 4) + ar;
      *(unsigned int*)(&As[r * LDSP + kp]) =
          pack2_bf16(va[i].x * a_scale, va[i].y * a_scale);
    }
#pragma unroll
    for (int i = 0; i < 4; ++i) {
      int cc = (i << 4) + ar;
      *(unsigned int*)(&Bs[cc * LDSP + kp]) = pack2_bf16(vb[i].x, vb[i].y);
    }
    __syncthreads();

    // ---- fragments ----
    // A: lane (g,L) holds row tile*16+L, K = {g*8..g*8+7} u {16+g*8..16+g*8+7}
    // B: lane (g,L) holds col tile*16+L, K = {g*8..g*8+7} u {16+g*8..16+g*8+7}
    union Frag { uiv4 q[2]; v16bf v; };
    Frag af[2], bf[2];
#pragma unroll
    for (int mi = 0; mi < 2; ++mi) {
      const int arow = (wm2 + mi) * 16 + L;
      af[mi].q[0] = *(const uiv4*)(&As[arow * LDSP + g * 8]);
      af[mi].q[1] = *(const uiv4*)(&As[arow * LDSP + 16 + g * 8]);
    }
#pragma unroll
    for (int ti = 0; ti < 2; ++ti) {
      const int bcol = (wn2 + ti) * 16 + L;
      bf[ti].q[0] = *(const uiv4*)(&Bs[bcol * LDSP + g * 8]);
      bf[ti].q[1] = *(const uiv4*)(&Bs[bcol * LDSP + 16 + g * 8]);
    }

#pragma unroll
    for (int mi = 0; mi < 2; ++mi)
#pragma unroll
      for (int ti = 0; ti < 2; ++ti)
        acc[mi][ti] = __builtin_amdgcn_wmma_f32_16x16x32_bf16(
            false, af[mi].v, false, bf[ti].v, (short)0, acc[mi][ti], false, false);

    __syncthreads();
  }

  // ---- epilogue. C/D layout: VGPR r of lane (g,L) -> (M = g*8 + r, N = L) ----
#pragma unroll
  for (int ti = 0; ti < 2; ++ti) {
    int col = n0 + (wn2 + ti) * 16 + L;
    if (!ALIGNED && col >= Nc) continue;
    float bcol = bias ? bias[col] : 0.0f;
#pragma unroll
    for (int mi = 0; mi < 2; ++mi) {
#pragma unroll
      for (int r = 0; r < 8; ++r) {
        int row = m0 + (wm2 + mi) * 16 + g * 8 + r;
        if (!ALIGNED && row >= Mr) continue;
        float val = acc[mi][ti][r] + bcol;
        if (rowdiv) val /= rowdiv[row];
        if (act)    val = gelu_exact(val);
        if (resid)  val += resid[(long)row * r_rs + col];
        C[(long)row * c_rs + col] = val;
      }
    }
  }
}

// ---------------------------------------------------------------------------
// Elementwise / reduction kernels
// ---------------------------------------------------------------------------

// h = x @ to_in_w.T + to_in_b + pos_emb
__global__ __launch_bounds__(256) void embed_kernel(
    const float* __restrict__ x, const float* __restrict__ w,
    const float* __restrict__ b, const float* __restrict__ pos,
    float* __restrict__ h)
{
  int idx = blockIdx.x * 256 + threadIdx.x;
  if (idx >= NTOK * DIMC) return;
  int c   = idx % DIMC;
  int tok = idx / DIMC;
  int p   = tok & (NPIX - 1);
  const float* xp = x + (long)tok * 3;
  h[idx] = b[c] + xp[0] * w[c * 3 + 0] + xp[1] * w[c * 3 + 1] + xp[2] * w[c * 3 + 2]
         + pos[(long)p * DIMC + c];
}

// ScaleNorm: y = x / max(||x|| * d^-0.5, eps) * g   (wave per row of 192)
__global__ __launch_bounds__(256) void scalenorm_kernel(
    const float* __restrict__ in, float* __restrict__ out,
    const float* __restrict__ gp, int rows)
{
  int row  = blockIdx.x * 8 + (threadIdx.x >> 5);
  int lane = threadIdx.x & 31;
  if (row >= rows) return;
  const float* p = in + (long)row * DIMC;
  float s = 0.0f;
  for (int c = lane; c < DIMC; c += 32) { float v = p[c]; s += v * v; }
  for (int off = 16; off; off >>= 1) s += __shfl_xor(s, off, 32);
  float n   = sqrtf(s) * 0.07216878364870f;   // 192^-0.5
  float scl = gp[0] / fmaxf(n, EPS_SN);
  for (int c = lane; c < DIMC; c += 32) out[(long)row * DIMC + c] = p[c] * scl;
}

// diag[n] = 0.5 * dn^2 * sum_d q[n,d]^2   (q slice: stride INNERC, 64 cols)
__global__ __launch_bounds__(256) void diag_kernel(
    const float* __restrict__ q, float* __restrict__ diag)
{
  int row  = blockIdx.x * 8 + (threadIdx.x >> 5);
  int lane = threadIdx.x & 31;
  const float* p = q + (long)row * INNERC;
  float v0 = p[lane], v1 = p[lane + 32];
  float s = v0 * v0 + v1 * v1;
  for (int off = 16; off; off >>= 1) s += __shfl_xor(s, off, 32);
  if (!lane) diag[row] = 0.5f * DN2_CONST * s;
}

// rmax[n] = max_m dd[n,m]
__global__ __launch_bounds__(256) void rowmax_kernel(
    const float* __restrict__ dd, float* __restrict__ rmax)
{
  int row  = blockIdx.x * 8 + (threadIdx.x >> 5);
  int lane = threadIdx.x & 31;
  const float* p = dd + (long)row * MLD;
  float s = -3.4e38f;
  for (int m = lane; m < MFEAT; m += 32) s = fmaxf(s, p[m]);
  for (int off = 16; off; off >>= 1) s = fmaxf(s, __shfl_xor(s, off, 32));
  if (!lane) rmax[row] = s;
}

// gmax[0] = max_n rmax[n]
__global__ __launch_bounds__(256) void gmax_kernel(
    const float* __restrict__ rmax, float* __restrict__ gmax)
{
  __shared__ float sm[256];
  float s = -3.4e38f;
  for (int i = threadIdx.x; i < NPIX; i += 256) s = fmaxf(s, rmax[i]);
  sm[threadIdx.x] = s;
  __syncthreads();
  for (int off = 128; off; off >>= 1) {
    if (threadIdx.x < (unsigned)off) sm[threadIdx.x] = fmaxf(sm[threadIdx.x], sm[threadIdx.x + off]);
    __syncthreads();
  }
  if (!threadIdx.x) gmax[0] = sm[0];
}

// dd[n,m] = ratio * (exp(dd - diag[n] - mx) + eps); mx per-row or global
__global__ __launch_bounds__(256) void feature_kernel(
    float* __restrict__ dd, const float* __restrict__ diag,
    const float* __restrict__ mx, int per_row)
{
  int idx = blockIdx.x * 256 + threadIdx.x;
  if (idx >= NPIX * MFEAT) return;
  int n = idx / MFEAT, m = idx % MFEAT;
  float mv = per_row ? mx[n] : mx[0];
  long  o  = (long)n * MLD + m;
  dd[o] = RATIO_CONST * (expf(dd[o] - diag[n] - mv) + EPS_K);
}

// ksum[m] = sum_n kp[n,m]   (one block per column)
__global__ __launch_bounds__(256) void colsum_kernel(
    const float* __restrict__ kp, float* __restrict__ ksum)
{
  int m = blockIdx.x;
  __shared__ float sm[256];
  float s = 0.0f;
  for (int n = threadIdx.x; n < NPIX; n += 256) s += kp[(long)n * MLD + m];
  sm[threadIdx.x] = s;
  __syncthreads();
  for (int off = 128; off; off >>= 1) {
    if (threadIdx.x < (unsigned)off) sm[threadIdx.x] += sm[threadIdx.x + off];
    __syncthreads();
  }
  if (!threadIdx.x) ksum[m] = sm[0];
}

// denom[n] = dot(qp[n,:], ksum)   (wave per row)
__global__ __launch_bounds__(256) void denom_kernel(
    const float* __restrict__ qp, const float* __restrict__ ksum,
    float* __restrict__ den)
{
  int row  = blockIdx.x * 8 + (threadIdx.x >> 5);
  int lane = threadIdx.x & 31;
  const float* p = qp + (long)row * MLD;
  float s = 0.0f;
  for (int m = lane; m < MFEAT; m += 32) s += p[m] * ksum[m];
  for (int off = 16; off; off >>= 1) s += __shfl_xor(s, off, 32);
  if (!lane) den[row] = s;
}

// [b,4096,192] -> [b,192,4096]
__global__ __launch_bounds__(256) void nhwc2nchw_kernel(
    const float* __restrict__ in, float* __restrict__ out)
{
  int idx = blockIdx.x * 256 + threadIdx.x;
  if (idx >= BATCH * DIMC * NPIX) return;
  int p = idx % NPIX;
  int c = (idx / NPIX) % DIMC;
  int b = idx / (NPIX * DIMC);
  out[idx] = in[((long)b * NPIX + p) * DIMC + c];
}

// GroupNorm over (24 ch x 4096 px) per (batch, group); 32 blocks
__global__ __launch_bounds__(256) void groupnorm_kernel(
    const float* __restrict__ in, float* __restrict__ out,
    const float* __restrict__ gw, const float* __restrict__ gb)
{
  const int batch = blockIdx.x >> 3, grp = blockIdx.x & 7;
  const long base = ((long)batch * DIMC + grp * 24) * NPIX;
  const int NE = 24 * NPIX;
  __shared__ float ssum[256], ssq[256];
  float s = 0.0f, q = 0.0f;
  for (int i = threadIdx.x; i < NE; i += 256) { float v = in[base + i]; s += v; q += v * v; }
  ssum[threadIdx.x] = s; ssq[threadIdx.x] = q;
  __syncthreads();
  for (int off = 128; off; off >>= 1) {
    if (threadIdx.x < (unsigned)off) {
      ssum[threadIdx.x] += ssum[threadIdx.x + off];
      ssq[threadIdx.x]  += ssq[threadIdx.x + off];
    }
    __syncthreads();
  }
  float mean = ssum[0] / (float)NE;
  float var  = ssq[0] / (float)NE - mean * mean;
  float inv  = rsqrtf(var + EPS_GN);
  for (int i = threadIdx.x; i < NE; i += 256) {
    int c = grp * 24 + i / NPIX;
    out[base + i] = (in[base + i] - mean) * inv * gw[c] + gb[c];
  }
}

// Grouped circular conv + exact GELU (64x64 grid, wrap via &63)
__global__ __launch_bounds__(256) void conv_gelu_kernel(
    const float* __restrict__ in, const float* __restrict__ w,
    const float* __restrict__ bias, float* __restrict__ out,
    int Cin, int Cout, int icpg, int ocpg, int ksz)
{
  int idx = blockIdx.x * 256 + threadIdx.x;
  int total = BATCH * Cout * NPIX;
  if (idx >= total) return;
  int pix = idx % NPIX;
  int oc  = (idx / NPIX) % Cout;
  int b   = idx / (NPIX * Cout);
  int x = pix & 63, y = pix >> 6;
  int grp = oc / ocpg;
  int p = ksz >> 1;
  float acc = bias[oc];
  const float* wp = w + (long)oc * icpg * ksz * ksz;
  const float* ip = in + ((long)b * Cin + grp * icpg) * NPIX;
  for (int ic = 0; ic < icpg; ++ic) {
    for (int ky = 0; ky < ksz; ++ky) {
      int yy = (y + ky - p + 64) & 63;
      const float* irow = ip + ic * NPIX + yy * 64;
      const float* wrow = wp + (ic * ksz + ky) * ksz;
      for (int kx = 0; kx < ksz; ++kx) {
        int xx = (x + kx - p + 64) & 63;
        acc += irow[xx] * wrow[kx];
      }
    }
  }
  out[idx] = gelu_exact(acc);
}

// Final 1x1 conv [48->3] + transpose to [b, ny*nx, 3] output layout
__global__ __launch_bounds__(256) void conv1_out_kernel(
    const float* __restrict__ in, const float* __restrict__ w,
    const float* __restrict__ bias, float* __restrict__ out)
{
  int idx = blockIdx.x * 256 + threadIdx.x;
  if (idx >= BATCH * NPIX * 3) return;
  int c   = idx % 3;
  int pix = (idx / 3) % NPIX;
  int b   = idx / (3 * NPIX);
  float acc = bias[c];
  const float* ip = in + (long)b * 48 * NPIX + pix;
#pragma unroll
  for (int ic = 0; ic < 48; ++ic) acc += ip[ic * NPIX] * w[c * 48 + ic];
  out[idx] = acc;
}

// ---------------------------------------------------------------------------
// Host orchestration
// ---------------------------------------------------------------------------
extern "C" void kernel_launch(void* const* d_in, const int* in_sizes, int n_in,
                              void* d_out, int out_size, void* d_ws, size_t ws_size,
                              hipStream_t stream) {
  const float* x        = (const float*)d_in[0];
  const float* to_in_w  = (const float*)d_in[1];
  const float* to_in_b  = (const float*)d_in[2];
  const float* pos_emb  = (const float*)d_in[3];
  const float* proj     = (const float*)d_in[4];
  const float* sn_attn_g= (const float*)d_in[5];
  const float* wq       = (const float*)d_in[6];
  const float* wk       = (const float*)d_in[7];
  const float* wv       = (const float*)d_in[8];
  const float* wo       = (const float*)d_in[9];
  const float* bo       = (const float*)d_in[10];
  const float* sn_ff_g  = (const float*)d_in[11];
  const float* w1       = (const float*)d_in[12];
  const float* b1       = (const float*)d_in[13];
  const float* w2       = (const float*)d_in[14];
  const float* b2       = (const float*)d_in[15];
  const float* expand_w = (const float*)d_in[16];
  const float* fwd_w    = (const float*)d_in[17];
  const float* dec_lin_w= (const float*)d_in[18];
  const float* dec_lin_b= (const float*)d_in[19];
  const float* gn_g     = (const float*)d_in[20];
  const float* gn_b     = (const float*)d_in[21];
  const float* c9_w     = (const float*)d_in[22];
  const float* c9_b     = (const float*)d_in[23];
  const float* c7_w     = (const float*)d_in[24];
  const float* c7_b     = (const float*)d_in[25];
  const float* c5_w     = (const float*)d_in[26];
  const float* c5_b     = (const float*)d_in[27];
  const float* c1_w     = (const float*)d_in[28];
  const float* c1_b     = (const float*)d_in[29];
  // d_in[30] = latent_steps (== 1, compile-time constant here)

  float* ws = (float*)d_ws;
  float* hB   = ws + 0;            // 16384*192  = 3,145,728
  float* hn   = ws + 3145728;      // 3,145,728
  float* qB   = ws + 6291456;      // 16384*512  = 8,388,608
  float* kB   = ws + 14680064;     // 8,388,608
  float* vB   = ws + 23068672;     // 8,388,608
  float* aoB  = ws + 31457280;     // 8,388,608
  float* qpB  = ws + 39845888;     // 4096*272   = 1,114,112
  float* kpB  = ws + 40960000;     // 1,114,112
  float* ctxB = ws + 42074112;     // 272*64
  float* ksumB= ws + 42091520;     // 512
  float* diagB= ws + 42092032;     // 4096
  float* rmaxB= ws + 42096128;     // 4096
  float* gmaxB= ws + 42100224;     // 64
  float* denB = ws + 42100288;     // 4096
  // phase-local aliases (regions dead by the time they are reused):
  float* ff1  = qB;                // 16384*768 fits in q..k region
  float* ybuf = qB;                // decode NCHW
  float* gbuf = qB + 3200000;      // groupnorm out (still inside q region)
  float* c9o  = vB;
  float* c7o  = aoB;
  float* c5o  = aoB + 2000000;

  auto gemm = [&](const float* A, long ars, long aks, float ascale,
                  const float* Bp, long bks, long bcs,
                  float* C, long crs, int Mr, int Nc, int K,
                  const float* bias, const float* resid, long rrs,
                  const float* rowdiv, int act) {
    dim3 grid((Nc + BNT - 1) / BNT, (Mr + BMT - 1) / BMT);
    bool aligned = (Mr % BMT == 0) && (Nc % BNT == 0) && (K % BKT == 0) &&
                   (aks == 1) && (bks == 1);
    if (aligned) {
      hipLaunchKernelGGL(wmma_gemm_kernel<true>, grid, dim3(256), 0, stream,
                         A, ars, aks, ascale, Bp, bks, bcs, C, crs, Mr, Nc, K,
                         bias, resid, rrs, rowdiv, act);
    } else {
      hipLaunchKernelGGL(wmma_gemm_kernel<false>, grid, dim3(256), 0, stream,
                         A, ars, aks, ascale, Bp, bks, bcs, C, crs, Mr, Nc, K,
                         bias, resid, rrs, rowdiv, act);
    }
  };

  // ---- embed: h = x @ to_in_w.T + b + pos_emb
  hipLaunchKernelGGL(embed_kernel, dim3((NTOK * DIMC + 255) / 256), dim3(256), 0, stream,
                     x, to_in_w, to_in_b, pos_emb, hB);

  for (int i = 0; i < DEPTHC; ++i) {
    // ---- attention (prenorm) ----
    hipLaunchKernelGGL(scalenorm_kernel, dim3(NTOK / 8), dim3(256), 0, stream,
                       hB, hn, sn_attn_g + i, NTOK);
    gemm(hn, DIMC, 1, 1.0f, wq + (long)i * INNERC * DIMC, 1, DIMC,
         qB, INNERC, NTOK, INNERC, DIMC, nullptr, nullptr, 0, nullptr, 0);
    gemm(hn, DIMC, 1, 1.0f, wk + (long)i * INNERC * DIMC, 1, DIMC,
         kB, INNERC, NTOK, INNERC, DIMC, nullptr, nullptr, 0, nullptr, 0);
    gemm(hn, DIMC, 1, 1.0f, wv + (long)i * INNERC * DIMC, 1, DIMC,
         vB, INNERC, NTOK, INNERC, DIMC, nullptr, nullptr, 0, nullptr, 0);

    const float* projL = proj + (long)i * MFEAT * DHC;
    for (int bh = 0; bh < BATCH * 8; ++bh) {
      int bb = bh >> 3, hh = bh & 7;
      const float* qs = qB + ((long)bb * NPIX) * INNERC + hh * DHC;
      const float* ks = kB + ((long)bb * NPIX) * INNERC + hh * DHC;
      const float* vs = vB + ((long)bb * NPIX) * INNERC + hh * DHC;
      float*       as = aoB + ((long)bb * NPIX) * INNERC + hh * DHC;

      // q features: dd = (q*dn) @ proj^T, then exp with per-row max
      gemm(qs, INNERC, 1, DN_CONST, projL, 1, DHC,
           qpB, MLD, NPIX, MFEAT, DHC, nullptr, nullptr, 0, nullptr, 0);
      hipLaunchKernelGGL(diag_kernel, dim3(NPIX / 8), dim3(256), 0, stream, qs, diagB);
      hipLaunchKernelGGL(rowmax_kernel, dim3(NPIX / 8), dim3(256), 0, stream, qpB, rmaxB);
      hipLaunchKernelGGL(feature_kernel, dim3((NPIX * MFEAT + 255) / 256), dim3(256), 0, stream,
                         qpB, diagB, rmaxB, 1);

      // k features: global max over (n, m)
      gemm(ks, INNERC, 1, DN_CONST, projL, 1, DHC,
           kpB, MLD, NPIX, MFEAT, DHC, nullptr, nullptr, 0, nullptr, 0);
      hipLaunchKernelGGL(diag_kernel, dim3(NPIX / 8), dim3(256), 0, stream, ks, diagB);
      hipLaunchKernelGGL(rowmax_kernel, dim3(NPIX / 8), dim3(256), 0, stream, kpB, rmaxB);
      hipLaunchKernelGGL(gmax_kernel, dim3(1), dim3(256), 0, stream, rmaxB, gmaxB);
      hipLaunchKernelGGL(feature_kernel, dim3((NPIX * MFEAT + 255) / 256), dim3(256), 0, stream,
                         kpB, diagB, gmaxB, 0);

      // context[m,d] = sum_n kp[n,m] * v[n,d]   (A = kp^T via strides)
      hipLaunchKernelGGL(colsum_kernel, dim3(MFEAT), dim3(256), 0, stream, kpB, ksumB);
      gemm(kpB, 1, MLD, 1.0f, vs, INNERC, 1,
           ctxB, DHC, MFEAT, DHC, NPIX, nullptr, nullptr, 0, nullptr, 0);
      // denom[n] = qp[n,:] . ksum
      hipLaunchKernelGGL(denom_kernel, dim3(NPIX / 8), dim3(256), 0, stream, qpB, ksumB, denB);
      // out = (qp @ context) / denom  -> ao slice
      gemm(qpB, MLD, 1, 1.0f, ctxB, DHC, 1,
           as, INNERC, NPIX, DHC, MFEAT, nullptr, nullptr, 0, denB, 0);
    }

    // output projection + residual: h = h + ao @ wo^T + bo
    gemm(aoB, INNERC, 1, 1.0f, wo + (long)i * DIMC * INNERC, 1, INNERC,
         hB, DIMC, NTOK, DIMC, INNERC, bo + (long)i * DIMC, hB, DIMC, nullptr, 0);

    // ---- feed-forward (prenorm) ----
    hipLaunchKernelGGL(scalenorm_kernel, dim3(NTOK / 8), dim3(256), 0, stream,
                       hB, hn, sn_ff_g + i, NTOK);
    gemm(hn, DIMC, 1, 1.0f, w1 + (long)i * FFC * DIMC, 1, DIMC,
         ff1, FFC, NTOK, FFC, DIMC, b1 + (long)i * FFC, nullptr, 0, nullptr, 1);
    gemm(ff1, FFC, 1, 1.0f, w2 + (long)i * DIMC * FFC, 1, FFC,
         hB, DIMC, NTOK, DIMC, FFC, b2 + (long)i * DIMC, hB, DIMC, nullptr, 0);
  }

  // ---- decode head ----
  // expand_latent (no bias), unitary forward (latent_steps = 1), dec_lin (+bias)
  gemm(hB, DIMC, 1, 1.0f, expand_w, 1, DIMC, hn, DIMC, NTOK, DIMC, DIMC,
       nullptr, nullptr, 0, nullptr, 0);
  gemm(hn, DIMC, 1, 1.0f, fwd_w, 1, DIMC, hB, DIMC, NTOK, DIMC, DIMC,
       nullptr, nullptr, 0, nullptr, 0);
  gemm(hB, DIMC, 1, 1.0f, dec_lin_w, 1, DIMC, hn, DIMC, NTOK, DIMC, DIMC,
       dec_lin_b, nullptr, 0, nullptr, 0);

  hipLaunchKernelGGL(nhwc2nchw_kernel, dim3((BATCH * DIMC * NPIX + 255) / 256), dim3(256), 0, stream,
                     hn, ybuf);
  hipLaunchKernelGGL(groupnorm_kernel, dim3(BATCH * 8), dim3(256), 0, stream,
                     ybuf, gbuf, gn_g, gn_b);

  hipLaunchKernelGGL(conv_gelu_kernel, dim3((BATCH * 192 * NPIX + 255) / 256), dim3(256), 0, stream,
                     gbuf, c9_w, c9_b, c9o, 192, 192, 24, 24, 9);
  hipLaunchKernelGGL(conv_gelu_kernel, dim3((BATCH * 96 * NPIX + 255) / 256), dim3(256), 0, stream,
                     c9o, c7_w, c7_b, c7o, 192, 96, 24, 12, 7);
  hipLaunchKernelGGL(conv_gelu_kernel, dim3((BATCH * 48 * NPIX + 255) / 256), dim3(256), 0, stream,
                     c7o, c5_w, c5_b, c5o, 96, 48, 12, 6, 5);
  hipLaunchKernelGGL(conv1_out_kernel, dim3((BATCH * NPIX * 3 + 255) / 256), dim3(256), 0, stream,
                     c5o, c1_w, c1_b, (float*)d_out);

  (void)in_sizes; (void)n_in; (void)out_size; (void)ws_size;
}
